// GptOssMLP_19550691131682
// MI455X (gfx1250) — matrix operand order repaired and checked by reading, not compiled, not run on hardware
//
#include <hip/hip_runtime.h>

// ---------- problem constants ----------
#define T_TOK   1024
#define H_DIM   2880
#define F_DIM   2880
#define TWOF    5760
#define NEXP    16
#define TOPK    4
#define KT      32      // K tile per WMMA loop step
#define KPAD    40      // padded K stride in LDS (bf16 elems), 80B rows keep 16B align
#define MBLK    128     // tokens per block (8 waves x 16-row M tiles)
#define NBLK    64      // output columns per block (4 x 16-col WMMA tiles)

typedef __attribute__((ext_vector_type(16))) __bf16 v16bf;
typedef __attribute__((ext_vector_type(8)))  float  v8f;
typedef __attribute__((ext_vector_type(4)))  unsigned int v4u;

union FragU { v16bf v; uint4 q[2]; };

static __device__ __forceinline__ unsigned short f2bf(float f) {
    unsigned u = __float_as_uint(f);
    u += 0x7fffu + ((u >> 16) & 1u);          // round-to-nearest-even
    return (unsigned short)(u >> 16);
}

static __device__ __forceinline__ v8f wmma_bf16(v16bf a, v16bf b, v8f c) {
    return __builtin_amdgcn_wmma_f32_16x16x32_bf16(
        /*neg_a=*/false, a, /*neg_b=*/false, b,
        /*c_mod=*/(short)0, c, /*reuse_a=*/false, /*reuse_b=*/false);
}

// Stage one KT x NBLK fp32 weight tile into LDS as bf16, transposed to [n][k].
// Vectorized: each of 256 threads does two nontemporal b128 loads (4 consecutive n).
static __device__ __forceinline__ void stage_tile(
        unsigned short* __restrict__ lbuf, const float* __restrict__ W,
        size_t wbase, int ldw, int k0, int tid) {
    int q = tid & 15;            // which group of 4 columns
    int kk = tid >> 4;           // 0..15
    int n = q * 4;
#pragma unroll
    for (int h = 0; h < 2; ++h) {
        int k = kk + h * 16;
        const v4u* gp = (const v4u*)&W[wbase + (size_t)(k0 + k) * ldw + n];
        v4u w4 = __builtin_nontemporal_load(gp);
        lbuf[(n + 0) * KPAD + k] = f2bf(__uint_as_float(w4.x));
        lbuf[(n + 1) * KPAD + k] = f2bf(__uint_as_float(w4.y));
        lbuf[(n + 2) * KPAD + k] = f2bf(__uint_as_float(w4.z));
        lbuf[(n + 3) * KPAD + k] = f2bf(__uint_as_float(w4.w));
    }
}

// ---------- kernel 0: zero out + counts, convert x -> bf16 ----------
__global__ void init_kernel(const float* __restrict__ x,
                            float* __restrict__ out,
                            unsigned short* __restrict__ xb,
                            int* __restrict__ counts, int total) {
    int stride = gridDim.x * blockDim.x;
    for (int i = blockIdx.x * blockDim.x + threadIdx.x; i < total; i += stride) {
        out[i] = 0.0f;
        xb[i]  = f2bf(x[i]);
    }
    int g = blockIdx.x * blockDim.x + threadIdx.x;
    if (g < NEXP) counts[g] = 0;
}

// ---------- kernel 1: router (one wave per token) ----------
__global__ __launch_bounds__(128) void router_kernel(
        const float* __restrict__ x, const float* __restrict__ rw,
        const float* __restrict__ rb, int* __restrict__ counts,
        int* __restrict__ etok, float* __restrict__ ewgt) {
    int wave = threadIdx.x >> 5, lane = threadIdx.x & 31;
    int t = blockIdx.x * 4 + wave;
    const float* xr = x + (size_t)t * H_DIM;

    float acc[NEXP];
#pragma unroll
    for (int e = 0; e < NEXP; ++e) acc[e] = 0.0f;

    for (int h = lane; h < H_DIM; h += 32) {
        float xv = xr[h];
#pragma unroll
        for (int e = 0; e < NEXP; ++e) acc[e] += xv * rw[e * H_DIM + h];
    }

    float logit[NEXP];
#pragma unroll
    for (int e = 0; e < NEXP; ++e) {
        float v = acc[e];
        for (int off = 16; off >= 1; off >>= 1) v += __shfl_xor(v, off, 32);
        logit[e] = v + rb[e];
    }

    unsigned used = 0;
    int   idx[TOPK];
    float val[TOPK];
    for (int j = 0; j < TOPK; ++j) {
        float best = -3.4e38f; int bi = 0;
#pragma unroll
        for (int e = 0; e < NEXP; ++e)
            if (!((used >> e) & 1u) && logit[e] > best) { best = logit[e]; bi = e; }
        idx[j] = bi; val[j] = best; used |= 1u << bi;
    }
    float mx = val[0], s = 0.0f, ex[TOPK];
    for (int j = 0; j < TOPK; ++j) { ex[j] = __expf(val[j] - mx); s += ex[j]; }

    if (lane == 0) {
        for (int j = 0; j < TOPK; ++j) {
            int slot = atomicAdd(&counts[idx[j]], 1);
            etok[idx[j] * T_TOK + slot] = t;
            ewgt[idx[j] * T_TOK + slot] = ex[j] / s;
        }
    }
}

// ---------- kernel 2: exclusive scan over 16 counts ----------
__global__ void scan_kernel(const int* __restrict__ counts, int* __restrict__ offsets) {
    if (threadIdx.x == 0 && blockIdx.x == 0) {
        int s = 0;
        for (int e = 0; e < NEXP; ++e) { offsets[e] = s; s += counts[e]; }
        offsets[NEXP] = s;
    }
}

// ---------- kernel 3: gate_up GEMM (bf16 WMMA) + GLU -> act (bf16) ----------
__global__ __launch_bounds__(256) void gemm1_kernel(
        const unsigned short* __restrict__ xb, const float* __restrict__ gup_w,
        const float* __restrict__ gup_b, const int* __restrict__ etok,
        const int* __restrict__ counts, const int* __restrict__ offsets,
        unsigned short* __restrict__ act) {
    int e  = blockIdx.z;
    int Mt = counts[e];
    int m0 = blockIdx.y * MBLK;
    if (m0 >= Mt) return;
    int n0 = blockIdx.x * NBLK;

    int tid = threadIdx.x, wave = tid >> 5, lane = tid & 31;
    int halfsel = (lane >> 4) & 1;
    int mt0  = m0 + wave * 16;
    int mrow = mt0 + (lane & 15);
    int mr_c = (mrow < Mt) ? mrow : m0;
    int tok  = etok[e * T_TOK + mr_c];
    const unsigned short* arow = xb + (size_t)tok * H_DIM;

    __shared__ alignas(128) unsigned short lw[2][NBLK * KPAD];
    v8f c[4] = {};
    const size_t wbase = (size_t)e * H_DIM * TWOF + n0;
    const int NK = H_DIM / KT;   // 90

    stage_tile(lw[0], gup_w, wbase, TWOF, 0, tid);
    __syncthreads();

    for (int kt = 0; kt < NK; ++kt) {
        int k0 = kt * KT;
        if (kt + 1 < NK)
            stage_tile(lw[(kt + 1) & 1], gup_w, wbase, TWOF, k0 + KT, tid);
        if (kt + 2 < NK && tid < 64) {   // prefetch tile k+2
            int k = tid >> 1, n = (tid & 1) * 32;
            __builtin_prefetch(&gup_w[wbase + (size_t)(k0 + 2 * KT + k) * TWOF + n], 0, 1);
        }

        const unsigned short* lb = lw[kt & 1];
        FragU a;
        int klo = k0 + halfsel * 8;
        a.q[0] = *(const uint4*)(arow + klo);
        a.q[1] = *(const uint4*)(arow + klo + 16);

        int kb = halfsel * 8;
        int nl = lane & 15;
#pragma unroll
        for (int f = 0; f < 4; ++f) {
            FragU b;
            b.q[0] = *(const uint4*)(&lb[(f * 16 + nl) * KPAD + kb]);
            b.q[1] = *(const uint4*)(&lb[(f * 16 + nl) * KPAD + kb + 16]);
            c[f] = wmma_bf16(a.v, b.v, c[f]);
        }
        __syncthreads();
    }

    int offs_e = offsets[e];
#pragma unroll
    for (int f = 0; f < 4; ++f) {
        int n_lane = n0 + f * 16 + (lane & 15);
        float bias = gup_b[e * TWOF + n_lane];
#pragma unroll
        for (int r = 0; r < 8; ++r) {
            int m = mt0 + r + halfsel * 8;
            float v = c[f][r] + bias;
            float p = __shfl_xor(v, 1, 32);       // pair gate/up columns
            float g = (lane & 1) ? p : v;
            float u = (lane & 1) ? v : p;
            g = fminf(g, 7.0f);
            u = fminf(fmaxf(u, -7.0f), 7.0f);
            float sig = 1.0f / (1.0f + __expf(-1.702f * g));
            float av  = (u + 1.0f) * (g * sig);
            if (((lane & 1) == 0) && m < Mt)
                act[(size_t)(offs_e + m) * F_DIM + (n_lane >> 1)] = f2bf(av);
        }
    }
}

// ---------- kernel 4: down GEMM (bf16 WMMA), scale + atomic combine ----------
__global__ __launch_bounds__(256) void gemm2_kernel(
        const unsigned short* __restrict__ act, const float* __restrict__ dw,
        const float* __restrict__ db, const int* __restrict__ etok,
        const float* __restrict__ ewgt, const int* __restrict__ counts,
        const int* __restrict__ offsets, float* __restrict__ out) {
    int e  = blockIdx.z;
    int Mt = counts[e];
    int m0 = blockIdx.y * MBLK;
    if (m0 >= Mt) return;
    int n0 = blockIdx.x * NBLK;

    int tid = threadIdx.x, wave = tid >> 5, lane = tid & 31;
    int halfsel = (lane >> 4) & 1;
    int mt0  = m0 + wave * 16;
    int mrow = mt0 + (lane & 15);
    int mr_c = (mrow < Mt) ? mrow : m0;
    int offs_e = offsets[e];
    const unsigned short* arow = act + (size_t)(offs_e + mr_c) * F_DIM;

    __shared__ alignas(128) unsigned short lw[2][NBLK * KPAD];
    v8f c[4] = {};
    const size_t wbase = (size_t)e * F_DIM * H_DIM + n0;
    const int NK = F_DIM / KT;   // 90

    stage_tile(lw[0], dw, wbase, H_DIM, 0, tid);
    __syncthreads();

    for (int kt = 0; kt < NK; ++kt) {
        int k0 = kt * KT;
        if (kt + 1 < NK)
            stage_tile(lw[(kt + 1) & 1], dw, wbase, H_DIM, k0 + KT, tid);
        if (kt + 2 < NK && tid < 64) {
            int k = tid >> 1, n = (tid & 1) * 32;
            __builtin_prefetch(&dw[wbase + (size_t)(k0 + 2 * KT + k) * H_DIM + n], 0, 1);
        }

        const unsigned short* lb = lw[kt & 1];
        FragU a;
        int klo = k0 + halfsel * 8;
        a.q[0] = *(const uint4*)(arow + klo);
        a.q[1] = *(const uint4*)(arow + klo + 16);

        int kb = halfsel * 8;
        int nl = lane & 15;
#pragma unroll
        for (int f = 0; f < 4; ++f) {
            FragU b;
            b.q[0] = *(const uint4*)(&lb[(f * 16 + nl) * KPAD + kb]);
            b.q[1] = *(const uint4*)(&lb[(f * 16 + nl) * KPAD + kb + 16]);
            c[f] = wmma_bf16(a.v, b.v, c[f]);
        }
        __syncthreads();
    }

    // hoist per-row routing metadata out of the f-loop
    int   toks[8];
    float wgts[8];
    bool  vld[8];
#pragma unroll
    for (int r = 0; r < 8; ++r) {
        int m  = mt0 + r + halfsel * 8;
        vld[r] = (m < Mt);
        int mc = vld[r] ? m : m0;
        toks[r] = etok[e * T_TOK + mc];
        wgts[r] = ewgt[e * T_TOK + mc];
    }

#pragma unroll
    for (int f = 0; f < 4; ++f) {
        int n_lane = n0 + f * 16 + (lane & 15);
        float bias = db[e * H_DIM + n_lane];
#pragma unroll
        for (int r = 0; r < 8; ++r) {
            if (vld[r])
                atomicAdd(&out[(size_t)toks[r] * H_DIM + n_lane],
                          wgts[r] * (c[f][r] + bias));
        }
    }
}

// ---------- host launch ----------
extern "C" void kernel_launch(void* const* d_in, const int* in_sizes, int n_in,
                              void* d_out, int out_size, void* d_ws, size_t ws_size,
                              hipStream_t stream) {
    const float* x    = (const float*)d_in[0];  // (1,1024,2880)
    const float* rw   = (const float*)d_in[1];  // (16,2880)
    const float* rb   = (const float*)d_in[2];  // (16,)
    const float* guw  = (const float*)d_in[3];  // (16,2880,5760)
    const float* gub  = (const float*)d_in[4];  // (16,5760)
    const float* dw   = (const float*)d_in[5];  // (16,2880,2880)
    const float* db   = (const float*)d_in[6];  // (16,2880)
    float* out = (float*)d_out;

    char* ws = (char*)d_ws;
    int*            counts  = (int*)ws;                                   // 16
    int*            offsets = counts + 16;                                // 17
    int*            etok    = (int*)(ws + 256);                           // 16*1024
    float*          ewgt    = (float*)(ws + 256 + 65536);                 // 16*1024
    unsigned short* xb      = (unsigned short*)(ws + 131328);             // T*H bf16
    unsigned short* actb    = (unsigned short*)(ws + 6029568);            // 4096*F bf16
    // total workspace use: ~29.6 MB

    init_kernel<<<2048, 256, 0, stream>>>(x, out, xb, counts, T_TOK * H_DIM);
    router_kernel<<<T_TOK / 4, 128, 0, stream>>>(x, rw, rb, counts, etok, ewgt);
    scan_kernel<<<1, 32, 0, stream>>>(counts, offsets);
    gemm1_kernel<<<dim3(TWOF / NBLK, T_TOK / MBLK, NEXP), 256, 0, stream>>>(
        xb, guw, gub, etok, counts, offsets, actb);
    gemm2_kernel<<<dim3(H_DIM / NBLK, T_TOK / MBLK, NEXP), 256, 0, stream>>>(
        actb, dw, db, etok, ewgt, counts, offsets, out);
}
